// DrBC_29033978921733
// MI455X (gfx1250) — matrix operand order repaired and checked by reading, not compiled, not run on hardware
//
#include <hip/hip_runtime.h>
#include <hip/hip_bf16.h>

typedef __attribute__((ext_vector_type(16))) __bf16 v16bf;
typedef __attribute__((ext_vector_type(8)))  float  v8f;
typedef __attribute__((ext_vector_type(4)))  unsigned int u32x4;
typedef __attribute__((ext_vector_type(8)))  int   i32x8;
typedef __attribute__((ext_vector_type(4)))  int   i32x4;

#define HDIM 128
#define H3   384
#define EPSN 1e-12f

__device__ __forceinline__ float sigmoidf(float x) { return 1.0f / (1.0f + __expf(-x)); }

// ---- Tensor Data Mover: DMA a rows x 128 f32 tile (row stride 128) to LDS.
// D# packing per CDNA5 ISA (2D tensor, groups 2/3 zero).
__device__ __forceinline__ void tdm_load_tile_f32(const float* gsrc, unsigned int lds_off,
                                                  int rows) {
    unsigned long long ga = (unsigned long long)(uintptr_t)gsrc;
    u32x4 g0;
    g0[0] = 1u;                                    // count=1, user descriptor, no gather
    g0[1] = lds_off;                               // lds_addr (bytes)
    g0[2] = (unsigned int)ga;                      // global_addr[31:0]
    g0[3] = (unsigned int)((ga >> 32) & 0x01FFFFFFu) | 0x80000000u; // addr[56:32] | type=2
    i32x8 g1;
    g1[0] = 0x00020000;                            // data_size=2 (4 bytes); mask/flags 0
    g1[1] = (HDIM << 16);                          // tensor_dim0 = 128  (bits 63:48)
    g1[2] = ((rows & 0xFFFF) << 16);               // tensor_dim1 = rows (bits 95:80)
    g1[3] = (HDIM << 16);                          // tile_dim0  = 128  (bits 127:112)
    g1[4] = (rows & 0xFFFF);                       // tile_dim1  = rows (bits 143:128)
    g1[5] = HDIM;                                  // tensor_dim0_stride = 128 (bits 191:160)
    g1[6] = 0;
    g1[7] = 0;
    i32x4 gz = {0, 0, 0, 0};
#if defined(__clang_major__) && (__clang_major__ >= 23)
    i32x8 gz8 = {0, 0, 0, 0, 0, 0, 0, 0};
    __builtin_amdgcn_tensor_load_to_lds(g0, g1, gz, gz, gz8, 0);
#else
    __builtin_amdgcn_tensor_load_to_lds(g0, g1, gz, gz, 0);
#endif
}

// ---------------------------------------------------------------- utilities
__global__ void fill_f32(float* __restrict__ p, float v, int count) {
    int i = blockIdx.x * blockDim.x + threadIdx.x;
    if (i < count) p[i] = v;
}

__global__ void cvt_bf16(const float* __restrict__ s, __bf16* __restrict__ d, int count) {
    int i = blockIdx.x * blockDim.x + threadIdx.x;
    if (i < count) d[i] = (__bf16)s[i];
}

__global__ void deg_scatter(const int* __restrict__ col, float* __restrict__ deg, int e) {
    int i = blockIdx.x * blockDim.x + threadIdx.x;
    if (i < e) atomicAdd(&deg[col[i]], 1.0f);
}

__global__ void deg_to_dinv(float* __restrict__ deg, int n) {
    int i = blockIdx.x * blockDim.x + threadIdx.x;
    if (i < n) deg[i] = rsqrtf(deg[i] + 1.0f);
}

__global__ void edge_norm(const int* __restrict__ row, const int* __restrict__ col,
                          const float* __restrict__ dinv, float* __restrict__ nrm, int e) {
    int i = blockIdx.x * blockDim.x + threadIdx.x;
    if (i < e) nrm[i] = dinv[row[i]] * dinv[col[i]];
}

// ------------------------------------------------------- encoder: x -> h0
__global__ __launch_bounds__(256) void encoder(const float* __restrict__ x,
                                               const float* __restrict__ We,
                                               const float* __restrict__ be,
                                               float* __restrict__ h,
                                               float* __restrict__ zmax, int n) {
    int wave = threadIdx.x >> 5, lane = threadIdx.x & 31;
    int node = blockIdx.x * 8 + wave;
    if (node >= n) return;
    float x0 = x[node * 3 + 0], x1 = x[node * 3 + 1], x2 = x[node * 3 + 2];
    float v[4]; float ss = 0.f;
#pragma unroll
    for (int q = 0; q < 4; ++q) {
        int j = lane + 32 * q;
        float y = be[j] + x0 * We[j * 3 + 0] + x1 * We[j * 3 + 1] + x2 * We[j * 3 + 2];
        y = fmaxf(y, 0.f);
        v[q] = y; ss += y * y;
    }
#pragma unroll
    for (int off = 16; off > 0; off >>= 1) ss += __shfl_xor(ss, off);
    float scale = 1.0f / fmaxf(sqrtf(ss), EPSN);
#pragma unroll
    for (int q = 0; q < 4; ++q) {
        int j = lane + 32 * q;
        float o = v[q] * scale;
        h[(size_t)node * HDIM + j]    = o;
        zmax[(size_t)node * HDIM + j] = o;   // init layer-wise max-pool with h0
    }
}

// --------------------------------------------- edge message scatter (agg)
__global__ __launch_bounds__(256) void msg_scatter(const int* __restrict__ row,
                                                   const int* __restrict__ col,
                                                   const float* __restrict__ nrm,
                                                   const float* __restrict__ h,
                                                   float* __restrict__ agg, int e) {
    long long idx = (long long)blockIdx.x * blockDim.x + threadIdx.x;
    int edge = (int)(idx >> 5);
    if (edge >= e) return;
    int lane = (int)idx & 31;
    int r = row[edge], c = col[edge];
    float w = nrm[edge];
    const float4 hv = *reinterpret_cast<const float4*>(h + (size_t)r * HDIM + lane * 4);
    float* dst = agg + (size_t)c * HDIM + lane * 4;
    atomicAdd(dst + 0, w * hv.x);
    atomicAdd(dst + 1, w * hv.y);
    atomicAdd(dst + 2, w * hv.z);
    atomicAdd(dst + 3, w * hv.w);
}

// ------------------------------------- fused GRU layer (WMMA bf16 -> f32)
// Block = 256 threads = 8 wave32s, one 16-node tile per block.
// Tile staging via TDM (tensor_load_to_lds -> f32 in sG scratch), converted
// to bf16 LDS fragments; gi/gh via v_wmma_f32_16x16x32_bf16; gate math +
// per-node L2-norm fused; h updated in place; zmax max-pooled.
__global__ __launch_bounds__(256) void gru_layer(const float* __restrict__ agg,
                                                 float* __restrict__ h,
                                                 const __bf16* __restrict__ wih,
                                                 const __bf16* __restrict__ whh,
                                                 const float* __restrict__ bih,
                                                 const float* __restrict__ bhh,
                                                 float* __restrict__ zmax, int n) {
    __shared__ __bf16 sA[16][HDIM];       // agg tile (bf16)      4 KB
    __shared__ __bf16 sH[16][HDIM];       // h   tile (bf16)      4 KB
    __shared__ float  sG[16][2 * H3];     // f32 staging, then gi||gh (48 KB)

    const int node0 = blockIdx.x * 16;
    const int tid = threadIdx.x;

    // --- stage agg & h tiles into LDS via the Tensor Data Mover (wave 0 issues)
    float* stage = &sG[0][0];                      // 16 KB of the sG region
    if (tid < 32) {
        unsigned int lds0 = (unsigned int)(uintptr_t)stage;
        tdm_load_tile_f32(agg + (size_t)node0 * HDIM, lds0, 16);
        tdm_load_tile_f32(h   + (size_t)node0 * HDIM, lds0 + 16 * HDIM * 4, 16);
        __builtin_amdgcn_s_wait_tensorcnt(0);      // DMA landed in LDS
    }
    __syncthreads();                               // publish staged tiles

    // --- convert staged f32 tiles to bf16 fragments
    for (int i = tid; i < 16 * HDIM; i += 256) {
        (&sA[0][0])[i] = (__bf16)stage[i];
        (&sH[0][0])[i] = (__bf16)stage[16 * HDIM + i];
    }
    __syncthreads();

    const int wave  = tid >> 5;
    const int lane  = tid & 31;
    const int mrow  = lane & 15;
    const int kbase = (lane >> 4) << 3;   // unused; A-frag takes its K half from khalf (0..15 vs 16..31)
    const int ncolL = lane & 15;
    const int khalf = (lane >> 4) << 4;   // B-frag: lane-half selects K 0..15 vs 16..31
    const int mbase = (lane >> 4) << 3;   // C/D: lane-half selects M rows 0..7 vs 8..15

    // 48 output tiles of 16 cols: t<24 -> gi (wih, A=agg); t>=24 -> gh (whh, A=h)
    for (int t = wave; t < 48; t += 8) {
        const bool is_ih = (t < 24);
        const int  ntile = is_ih ? t : (t - 24);
        const __bf16* __restrict__ W = is_ih ? wih : whh;
        __bf16 (*sX)[HDIM] = is_ih ? sA : sH;
        const int ncol = ntile * 16 + ncolL;
        const __bf16* wbase = W + (size_t)ncol * HDIM;
        // speculative prefetch of next tile's weight rows (dropped if OOB)
        __builtin_prefetch(wbase + (size_t)128 * HDIM, 0, 1);

        v8f acc = {};
#pragma unroll
        for (int kb = 0; kb < 4; ++kb) {          // K = 128 in 4 steps of 32
            const int k0 = kb * 32;
            v16bf afrag, bfrag;
#pragma unroll
            for (int j = 0; j < 16; ++j) {        // A: lane-half h holds K 16h..16h+15 (matches B)
                int k = k0 + khalf + j;
                afrag[j] = sX[mrow][k];
            }
            const __bf16* wrow = wbase + k0 + khalf;
#pragma unroll
            for (int j = 0; j < 16; ++j) bfrag[j] = wrow[j];   // B[k][n] = W[n][k]
            acc = __builtin_amdgcn_wmma_f32_16x16x32_bf16(false, afrag, false, bfrag,
                                                          (short)0, acc, false, false);
        }
        const int gcol = (is_ih ? 0 : H3) + ntile * 16 + ncolL;
#pragma unroll
        for (int i = 0; i < 8; ++i) sG[mbase + i][gcol] = acc[i];
    }
    __syncthreads();

    // gate fusion + L2 norm: wave w handles node rows 2w, 2w+1; 4 channels/lane
#pragma unroll
    for (int rr = 0; rr < 2; ++rr) {
        const int m = wave * 2 + rr;
        const int node = node0 + m;
        if (node >= n) continue;
        float val[4]; float ss = 0.f;
#pragma unroll
        for (int q = 0; q < 4; ++q) {
            const int c = lane + 32 * q;
            float gir = sG[m][c]            + bih[c];
            float giz = sG[m][HDIM + c]     + bih[HDIM + c];
            float gin = sG[m][2 * HDIM + c] + bih[2 * HDIM + c];
            float ghr = sG[m][H3 + c]            + bhh[c];
            float ghz = sG[m][H3 + HDIM + c]     + bhh[HDIM + c];
            float ghn = sG[m][H3 + 2 * HDIM + c] + bhh[2 * HDIM + c];
            float r  = sigmoidf(gir + ghr);
            float zg = sigmoidf(giz + ghz);
            float ng = tanhf(gin + r * ghn);
            float hv = h[(size_t)node * HDIM + c];
            float v  = (1.f - zg) * ng + zg * hv;
            val[q] = v; ss += v * v;
        }
#pragma unroll
        for (int off = 16; off > 0; off >>= 1) ss += __shfl_xor(ss, off);
        float scale = 1.f / fmaxf(sqrtf(ss), EPSN);
#pragma unroll
        for (int q = 0; q < 4; ++q) {
            const int c = lane + 32 * q;
            float o = val[q] * scale;
            h[(size_t)node * HDIM + c] = o;
            zmax[(size_t)node * HDIM + c] = fmaxf(zmax[(size_t)node * HDIM + c], o);
        }
    }
}

// ------------------------------------------------ decoder: zmax -> out[N]
__global__ __launch_bounds__(256) void decoder(const float* __restrict__ z,
                                               const float* __restrict__ Wh,
                                               const float* __restrict__ bh,
                                               const float* __restrict__ Wo,
                                               const float* __restrict__ bo,
                                               float* __restrict__ out, int n) {
    int wave = threadIdx.x >> 5, lane = threadIdx.x & 31;
    int node = blockIdx.x * 8 + wave;
    if (node >= n) return;
    const float* zr = z + (size_t)node * HDIM;
    float a0 = bh[lane], a1 = bh[lane + 32];
    const float* w0 = Wh + (size_t)lane * HDIM;
    const float* w1 = Wh + (size_t)(lane + 32) * HDIM;
    for (int k = 0; k < HDIM; k += 4) {
        float4 zv = *reinterpret_cast<const float4*>(zr + k);
        float4 wa = *reinterpret_cast<const float4*>(w0 + k);
        float4 wb = *reinterpret_cast<const float4*>(w1 + k);
        a0 += zv.x * wa.x + zv.y * wa.y + zv.z * wa.z + zv.w * wa.w;
        a1 += zv.x * wb.x + zv.y * wb.y + zv.z * wb.z + zv.w * wb.w;
    }
    a0 = fmaxf(a0, 0.f); a1 = fmaxf(a1, 0.f);
    float p = a0 * Wo[lane] + a1 * Wo[lane + 32];
#pragma unroll
    for (int off = 16; off > 0; off >>= 1) p += __shfl_xor(p, off);
    if (lane == 0) out[node] = p + bo[0];
}

// ------------------------------------------------------------------ launch
extern "C" void kernel_launch(void* const* d_in, const int* in_sizes, int n_in,
                              void* d_out, int out_size, void* d_ws, size_t ws_size,
                              hipStream_t stream) {
    const float* x     = (const float*)d_in[0];
    const int*   eidx  = (const int*)d_in[1];
    const float* We    = (const float*)d_in[2];
    const float* be    = (const float*)d_in[3];
    const float* w_ih  = (const float*)d_in[4];
    const float* w_hh  = (const float*)d_in[5];
    const float* b_ih  = (const float*)d_in[6];
    const float* b_hh  = (const float*)d_in[7];
    const float* W_hid = (const float*)d_in[8];
    const float* b_hid = (const float*)d_in[9];
    const float* W_out = (const float*)d_in[10];
    const float* b_out = (const float*)d_in[11];

    const int n = in_sizes[0] / 3;                     // 50000
    const int e = in_sizes[1] / 2;                     // 600000
    const int L = in_sizes[4] / (3 * HDIM * HDIM);     // 5
    const int* row = eidx;
    const int* col = eidx + e;

    size_t off = 0;
    auto carve = [&](size_t bytes) -> char* {
        char* p = (char*)d_ws + off;
        off += (bytes + 255) & ~(size_t)255;
        return p;
    };
    float*  deg  = (float*)carve((size_t)n * 4);
    float*  nrm  = (float*)carve((size_t)e * 4);
    float*  h    = (float*)carve((size_t)n * HDIM * 4);
    float*  agg  = (float*)carve((size_t)n * HDIM * 4);
    float*  zmax = (float*)carve((size_t)n * HDIM * 4);
    __bf16* wihb = (__bf16*)carve((size_t)L * H3 * HDIM * 2);
    __bf16* whhb = (__bf16*)carve((size_t)L * H3 * HDIM * 2);
    (void)ws_size; (void)n_in;

    const int wcount = L * H3 * HDIM;
    cvt_bf16<<<(wcount + 255) / 256, 256, 0, stream>>>(w_ih, wihb, wcount);
    cvt_bf16<<<(wcount + 255) / 256, 256, 0, stream>>>(w_hh, whhb, wcount);

    fill_f32<<<(n + 255) / 256, 256, 0, stream>>>(deg, 0.f, n);
    deg_scatter<<<(e + 255) / 256, 256, 0, stream>>>(col, deg, e);
    deg_to_dinv<<<(n + 255) / 256, 256, 0, stream>>>(deg, n);
    edge_norm<<<(e + 255) / 256, 256, 0, stream>>>(row, col, deg, nrm, e);

    encoder<<<(n + 7) / 8, 256, 0, stream>>>(x, We, be, h, zmax, n);

    const int nh = n * HDIM;
    const long long sthreads = (long long)e * 32;
    const int sblocks = (int)((sthreads + 255) / 256);
    const int gblocks = (n + 15) / 16;
    for (int l = 0; l < L; ++l) {
        fill_f32<<<(nh + 255) / 256, 256, 0, stream>>>(agg, 0.f, nh);
        msg_scatter<<<sblocks, 256, 0, stream>>>(row, col, nrm, h, agg, e);
        gru_layer<<<gblocks, 256, 0, stream>>>(agg, h,
            wihb + (size_t)l * H3 * HDIM, whhb + (size_t)l * H3 * HDIM,
            b_ih + (size_t)l * H3, b_hh + (size_t)l * H3, zmax, n);
    }

    decoder<<<(n + 7) / 8, 256, 0, stream>>>(zmax, W_hid, b_hid, W_out, b_out,
                                             (float*)d_out, n);
}
